// BudgetAwareHalting_16793322127882
// MI455X (gfx1250) — compile-verified, matrix operand order checked by reading, not run on hardware
//
#include <hip/hip_runtime.h>
#include <hip/hip_bf16.h>

// ---- problem constants (from reference) ----
#define Hdim 1024
#define Ddim 512
#define Bdim 16
#define Sdim 4096
#define Mtot (Bdim * Sdim)   // 65536 rows of flattened states

// ---- GEMM tiling ----
#define BM 128
#define BN 128
#define BK 32

typedef __attribute__((ext_vector_type(16))) __bf16 v16bf;
typedef __attribute__((ext_vector_type(8)))  __bf16 v8bf;
typedef __attribute__((ext_vector_type(8)))  float  v8f;
typedef __attribute__((ext_vector_type(4)))  float  f4;   // raw clang vector (nontemporal-compatible)

// =====================================================================
// Kernel 0: zero the comp-accumulator region of workspace
// =====================================================================
__global__ void zero_kernel(float* __restrict__ p, int n) {
    int i = blockIdx.x * blockDim.x + threadIdx.x;
    if (i < n) p[i] = 0.0f;
}

// =====================================================================
// Kernel 1: fused dual GEMM + output copy.
//   A = states [M=65536, K=1024] (f32 -> bf16 on the fly)
//   N-blocks 0..3: W = hW1[:H]  -> write pre = A@W + hb1 (bf16) to d_ws
//   N-blocks 4..7: W = cW1      -> h = relu(A@W + cb1); zacc[m] += h . cW2
//   N-block 0 additionally streams its A tiles to out (output == states),
//   with nontemporal stores so the copy does not evict L2-resident A tiles.
// Block: 256 threads (8 waves). Block tile 128x128, K-step 32.
// =====================================================================
__global__ __launch_bounds__(256)
void fused_gemm_kernel(const float* __restrict__ states,
                       const float* __restrict__ hW1,
                       const float* __restrict__ hb1,
                       const float* __restrict__ cW1,
                       const float* __restrict__ cb1,
                       const float* __restrict__ cW2,
                       __bf16* __restrict__ pre_ws,  // [M, 512] bf16
                       float* __restrict__ zacc,     // [M]
                       float* __restrict__ outp)     // [M, H] copy of states
{
    __shared__ __bf16 As[BM * BK];   // row-major [m][k], 64B rows
    __shared__ __bf16 Bs[BN * BK];   // transposed [n][k], 64B rows

    const int tid  = threadIdx.x;
    const int wave = tid >> 5;
    const int lane = tid & 31;
    const int half = lane >> 4;      // ISA 16-bit A layout: K chunk select
    const int l15  = lane & 15;

    const int m0   = blockIdx.y * BM;        // y = M dimension (512 blocks)
    const int nblk = blockIdx.x;             // x = N dimension (8 blocks, fast
                                             //     so A tile is reused via L2)
    const int n0g  = nblk * BN;
    const bool isPre  = (n0g < Ddim);
    const bool doCopy = (nblk == 0);
    const float* __restrict__ Wsrc = isPre ? hW1 : cW1;
    const float* __restrict__ bias = isPre ? hb1 : cb1;
    const int ncol0 = isPre ? n0g : (n0g - Ddim);

    v8f acc[8] = {};

    for (int k0 = 0; k0 < Hdim; k0 += BK) {
        // ---- cooperative load A tile: 128x32 f32 -> bf16 (+fused copy) ----
        #pragma unroll
        for (int it = 0; it < 4; ++it) {
            int idx = tid + it * 256;          // float4 slot 0..1023
            int r   = idx >> 3;                // row 0..127
            int c4  = idx & 7;                 // float4 within row
            size_t goff = (size_t)(m0 + r) * Hdim + k0 + c4 * 4;
            f4 v = *(const f4*)(states + goff);
            if (doCopy) {
                __builtin_nontemporal_store(v, (f4*)(outp + goff));
            }
            __bf16* dst = &As[r * BK + c4 * 4];
            dst[0] = (__bf16)v.x; dst[1] = (__bf16)v.y;
            dst[2] = (__bf16)v.z; dst[3] = (__bf16)v.w;
        }
        // ---- cooperative load B tile: 32x128 f32 -> bf16, transposed ----
        #pragma unroll
        for (int it = 0; it < 4; ++it) {
            int idx = tid + it * 256;          // 0..1023
            int kr  = idx >> 5;                // k row 0..31
            int c4  = idx & 31;                // float4 col group (128 cols / 4)
            f4 v = *(const f4*)(Wsrc + (size_t)(k0 + kr) * Ddim + ncol0 + c4 * 4);
            int n = c4 * 4;
            Bs[(n + 0) * BK + kr] = (__bf16)v.x;
            Bs[(n + 1) * BK + kr] = (__bf16)v.y;
            Bs[(n + 2) * BK + kr] = (__bf16)v.z;
            Bs[(n + 3) * BK + kr] = (__bf16)v.w;
        }
        __syncthreads();

        // prefetch next A K-slab (global_prefetch_b8)
        if (k0 + BK < Hdim) {
            __builtin_prefetch(states + (size_t)(m0 + (tid >> 1)) * Hdim
                                      + k0 + BK + (tid & 1) * 16, 0, 0);
        }

        // ---- A fragment (ISA 16-bit A 16x32 layout: two 16B chunks/lane) ----
        v8bf alo = *(v8bf*)&As[(wave * 16 + l15) * BK + half * 8];
        v8bf ahi = *(v8bf*)&As[(wave * 16 + l15) * BK + 16 + half * 8];
        v16bf a = __builtin_shufflevector(alo, ahi,
                   0,1,2,3,4,5,6,7,8,9,10,11,12,13,14,15);

        #pragma unroll
        for (int j = 0; j < 8; ++j) {
            v8bf blo = *(v8bf*)&Bs[(j * 16 + l15) * BK + half * 8];
            v8bf bhi = *(v8bf*)&Bs[(j * 16 + l15) * BK + 16 + half * 8];
            v16bf b = __builtin_shufflevector(blo, bhi,
                       0,1,2,3,4,5,6,7,8,9,10,11,12,13,14,15);
            acc[j] = __builtin_amdgcn_wmma_f32_16x16x32_bf16(
                         false, a, false, b, (short)0, acc[j], false, false);
        }
        __syncthreads();
    }

    // ---- epilogue ----
    // C/D layout: VGPR i, lane L -> M = i + (L<16 ? 0 : 8), N = L&15
    if (isPre) {
        #pragma unroll
        for (int j = 0; j < 8; ++j) {
            int n = n0g + j * 16 + l15;            // < 512
            float bv = bias[n];
            #pragma unroll
            for (int i = 0; i < 8; ++i) {
                int m = m0 + wave * 16 + i + half * 8;
                pre_ws[(size_t)m * Ddim + n] = (__bf16)(acc[j][i] + bv);
            }
        }
    } else {
        float psum[8];
        #pragma unroll
        for (int i = 0; i < 8; ++i) psum[i] = 0.0f;
        #pragma unroll
        for (int j = 0; j < 8; ++j) {
            int nc = ncol0 + j * 16 + l15;         // col within D
            float bv = bias[nc];
            float w2 = cW2[nc];
            #pragma unroll
            for (int i = 0; i < 8; ++i) {
                float h = acc[j][i] + bv;
                h = h > 0.0f ? h : 0.0f;
                psum[i] += h * w2;
            }
        }
        #pragma unroll
        for (int i = 0; i < 8; ++i) {
            float v = psum[i];
            v += __shfl_xor(v, 1);
            v += __shfl_xor(v, 2);
            v += __shfl_xor(v, 4);
            v += __shfl_xor(v, 8);
            if (l15 == 0) {
                int m = m0 + wave * 16 + i + half * 8;
                atomicAdd(&zacc[m], v);
            }
        }
    }
}

// =====================================================================
// Kernel 2: sequential budget scan (rb is uniform across B => scalar
// recurrence). One workgroup, 512 threads (one per d), 16 waves.
// Double-buffered: step t+1's 16x512 bf16 tile is fetched with
// global_load_async_to_lds_b128 (ASYNCcnt) while step t computes from LDS.
// =====================================================================
__global__ __launch_bounds__(512)
void scan_kernel(const __bf16* __restrict__ pre_ws,  // [M, 512] bf16
                 const float* __restrict__ hW1,      // w_budget row = hW1[H]
                 const float* __restrict__ hW2,
                 const float* __restrict__ hb2,
                 float* __restrict__ steps_out,      // [S] (as float)
                 float* __restrict__ rb_out)         // [1]
{
    __shared__ __bf16 Ls[2][Bdim * Ddim];   // 2 x 16KB double buffer
    __shared__ float sums[Bdim];
    __shared__ float sh_rb;
    __shared__ float sh_bin;

    const int tid  = threadIdx.x;                  // == d
    const int lane = tid & 31;
    const float wbd = hW1[(size_t)Hdim * Ddim + tid];   // w_budget[d]
    const float w2d = hW2[tid];
    const float b2  = hb2[0];

    // each thread owns two 16B chunks of the 16KB tile (1024 chunks total)
    const int c0 = tid;
    const int c1 = tid + 512;
    const unsigned lbase0 = (unsigned)(size_t)(&Ls[0][0]);
    const unsigned lbase1 = (unsigned)(size_t)(&Ls[1][0]);

    if (tid == 0) sh_rb = 128.0f;                  // TOTAL_BUDGET

    // issue async tile load for t = 0 into buffer 0
    {
        const __bf16* g0 = pre_ws + ((size_t)(c0 >> 6) * Sdim + 0) * Ddim + (c0 & 63) * 8;
        const __bf16* g1 = pre_ws + ((size_t)(c1 >> 6) * Sdim + 0) * Ddim + (c1 & 63) * 8;
        unsigned l0 = lbase0 + c0 * 16;
        unsigned l1 = lbase0 + c1 * 16;
        asm volatile("global_load_async_to_lds_b128 %0, %1, off\n\t"
                     "global_load_async_to_lds_b128 %2, %3, off"
                     :: "v"(l0), "v"(g0), "v"(l1), "v"(g1) : "memory");
    }

    for (int t = 0; t < Sdim; ++t) {
        // issue async tile load for t+1 into the other buffer (that buffer
        // was consumed at t-1; the end-of-iteration barrier made it free)
        {
            int tn = (t + 1 < Sdim) ? (t + 1) : (Sdim - 1);
            unsigned lb = ((t + 1) & 1) ? lbase1 : lbase0;
            const __bf16* g0 = pre_ws + ((size_t)(c0 >> 6) * Sdim + tn) * Ddim + (c0 & 63) * 8;
            const __bf16* g1 = pre_ws + ((size_t)(c1 >> 6) * Sdim + tn) * Ddim + (c1 & 63) * 8;
            unsigned l0 = lb + c0 * 16;
            unsigned l1 = lb + c1 * 16;
            asm volatile("global_load_async_to_lds_b128 %0, %1, off\n\t"
                         "global_load_async_to_lds_b128 %2, %3, off"
                         :: "v"(l0), "v"(g0), "v"(l1), "v"(g1) : "memory");
        }
        // wait for the older tile (t) only; keep the 2 newer loads in flight
        asm volatile("s_wait_asynccnt 0x2" ::: "memory");

        if (tid < Bdim) sums[tid] = 0.0f;
        if (tid == 0)   sh_bin = sh_rb * (1.0f / 128.0f);
        __syncthreads();

        const __bf16* Lbuf = Ls[t & 1];
        const float addv = sh_bin * wbd;
        #pragma unroll 4
        for (int b = 0; b < Bdim; ++b) {
            float v = (float)Lbuf[b * Ddim + tid] + addv;
            v = v > 0.0f ? v : 0.0f;
            float p = v * w2d;
            p += __shfl_xor(p, 1);
            p += __shfl_xor(p, 2);
            p += __shfl_xor(p, 4);
            p += __shfl_xor(p, 8);
            p += __shfl_xor(p, 16);
            if (lane == 0) atomicAdd(&sums[b], p);
        }
        __syncthreads();

        if (tid == 0) {
            float hm = 0.0f;
            #pragma unroll
            for (int b = 0; b < Bdim; ++b)
                hm += 1.0f / (1.0f + __expf(-(sums[b] + b2)));
            hm *= (1.0f / (float)Bdim);
            float rb = sh_rb;
            float bf = rb / (float)(Sdim - t + 1) / 8.0f;
            int maxfor = (int)truncf(bf * 8.0f);
            maxfor = maxfor < 1 ? 1 : (maxfor > 8 ? 8 : maxfor);
            int steps = (hm > 0.5f) ? 1 : maxfor;
            sh_rb = rb - (float)steps;
            steps_out[t] = (float)steps;
        }
        __syncthreads();
    }
    if (tid == 0) rb_out[0] = sh_rb;   // rb_final uniform -> its mean
}

// =====================================================================
// Kernel 3: comp_mean[s] = mean_b sigmoid(zacc[b*S+s] + cb2)
// =====================================================================
__global__ void comp_mean_kernel(const float* __restrict__ zacc,
                                 const float* __restrict__ cb2,
                                 float* __restrict__ out) {
    int s = blockIdx.x * blockDim.x + threadIdx.x;
    if (s >= Sdim) return;
    float c2 = cb2[0];
    float sum = 0.0f;
    #pragma unroll 4
    for (int b = 0; b < Bdim; ++b) {
        float z = zacc[(size_t)b * Sdim + s] + c2;
        sum += 1.0f / (1.0f + __expf(-z));
    }
    out[s] = sum * (1.0f / (float)Bdim);
}

// =====================================================================
extern "C" void kernel_launch(void* const* d_in, const int* in_sizes, int n_in,
                              void* d_out, int out_size, void* d_ws, size_t ws_size,
                              hipStream_t stream) {
    const float* states = (const float*)d_in[0];
    const float* hW1    = (const float*)d_in[1];   // [(H+1), D]
    const float* hb1    = (const float*)d_in[2];   // [D]
    const float* hW2    = (const float*)d_in[3];   // [D]
    const float* hb2    = (const float*)d_in[4];   // [1]
    const float* cW1    = (const float*)d_in[5];   // [H, D]
    const float* cb1    = (const float*)d_in[6];   // [D]
    const float* cW2    = (const float*)d_in[7];   // [D]
    const float* cb2    = (const float*)d_in[8];   // [1]

    float* out = (float*)d_out;
    // workspace layout: pre [M*512] bf16, then zacc [M] f32
    __bf16* pre_ws = (__bf16*)d_ws;
    float*  zacc   = (float*)((char*)d_ws + (size_t)Mtot * Ddim * sizeof(__bf16));

    // output layout (return order): states copy [M*H], steps_used [S],
    // comp_mean [S], rb_final_mean [1]
    const size_t out_states = (size_t)Mtot * Hdim;
    float* out_steps = out + out_states;
    float* out_comp  = out + out_states + Sdim;
    float* out_rb    = out + out_states + 2 * Sdim;

    zero_kernel<<<Mtot / 256, 256, 0, stream>>>(zacc, Mtot);

    dim3 grid(2 * Ddim / BN, Mtot / BM);   // (8, 512): N fast so A tile hits L2
    fused_gemm_kernel<<<grid, 256, 0, stream>>>(states, hW1, hb1, cW1, cb1, cW2,
                                                pre_ws, zacc, out);

    scan_kernel<<<1, 512, 0, stream>>>(pre_ws, hW1, hW2, hb2, out_steps, out_rb);

    comp_mean_kernel<<<Sdim / 256, 256, 0, stream>>>(zacc, cb2, out_comp);
}
